// CausalWanSelfAttention_26285199851919
// MI455X (gfx1250) — matrix operand order repaired and checked by reading, not compiled, not run on hardware
//
#include <hip/hip_runtime.h>
#include <hip/hip_bf16.h>

typedef __attribute__((ext_vector_type(16))) _Float16 v16h;
typedef __attribute__((ext_vector_type(8)))  _Float16 v8h;
typedef __attribute__((ext_vector_type(8)))  float    v8f;

#define S_TOT 2080
#define SPAD  2176
#define DIM   1536
#define NH    12
#define HD    128
#define ATT_SCALE 0.08838834764831845f   // 1/sqrt(128)

union V16 { v16h v; v8h h[2]; };

static __device__ __forceinline__ v8f wmma_f16(v16h a, v16h b, v8f c) {
  // D = A(16x32 f16) * B(32x16 f16) + C(16x16 f32)
  return __builtin_amdgcn_wmma_f32_16x16x32_f16(false, a, false, b, (short)0, c,
                                                false, false);
}

// ---------------------------------------------------------------------------
// f32 -> f16 conversion
// ---------------------------------------------------------------------------
__global__ void f32_to_f16_kernel(const float* __restrict__ src,
                                  _Float16* __restrict__ dst, int n) {
  int i = blockIdx.x * blockDim.x + threadIdx.x;
  if (i < n) dst[i] = (_Float16)src[i];
}

// ---------------------------------------------------------------------------
// GEMM: C[M x 1536] (f32) = A[M x 1536] (f16 row major) @ B[1536 x 1536]
// (f16 row major) + bias.
// Workgroup tile: 64M x 64N, 4 waves splitting M (16 rows each).
// The shared 32x64 B k-slice is staged into LDS with CDNA5 async
// global->LDS copies (ASYNCcnt), double buffered; A streams from global.
// grid = (ceil(M/64), 1536/64)
// ---------------------------------------------------------------------------
__global__ __launch_bounds__(128) void gemm_f16_kernel(
    const _Float16* __restrict__ A, const _Float16* __restrict__ B,
    const float* __restrict__ bias, float* __restrict__ C, int M) {
  __shared__ __align__(32) _Float16 sB[2][32][64];

  const int tid  = threadIdx.x;
  const int wave = tid >> 5;
  const int lane = tid & 31;
  const int g    = lane >> 4;    // lane group (0/1)
  const int ln   = lane & 15;
  const int m0   = blockIdx.x * 64;
  const int n0   = blockIdx.y * 64;

  // async-stage one 32x64 f16 B tile (rows k0..k0+31, cols n0..n0+63) into
  // sB[buf]; 256 16-byte chunks, 2 per thread -> 2 async instrs per wave.
  auto stage = [&](int k0, int buf) {
#pragma unroll
    for (int i = 0; i < 2; ++i) {
      const int c   = tid + 128 * i;    // chunk id 0..255
      const int row = c >> 3;           // 0..31   (K within slice)
      const int ch  = c & 7;            // 0..7    (16B chunk within row)
      const _Float16* gp = B + (size_t)(k0 + row) * DIM + n0 + ch * 8;
      unsigned lds = (unsigned)(uintptr_t)&sB[buf][row][ch * 8];
      unsigned long long ga = (unsigned long long)(uintptr_t)gp;
      asm volatile("global_load_async_to_lds_b128 %0, %1, off"
                   :: "v"(lds), "v"(ga) : "memory");
    }
  };

  const v8f vzero = {0.f, 0.f, 0.f, 0.f, 0.f, 0.f, 0.f, 0.f};
  v8f acc[4] = {vzero, vzero, vzero, vzero};

  const int arow = m0 + 16 * wave + ln;
  const _Float16* abase =
      A + (size_t)(arow < M ? arow : (M - 1)) * DIM;  // clamp tail reads

  stage(0, 0);

  const int KSTEPS = DIM / 32;  // 48
  for (int ks = 0; ks < KSTEPS; ++ks) {
    const int k0 = ks * 32;
    const int cb = ks & 1;
    if (ks + 1 < KSTEPS) {
      stage(k0 + 32, (ks + 1) & 1);
      // 4 async instrs outstanding per wave; <=2 left means stage(ks) done
      asm volatile("s_wait_asynccnt 2" ::: "memory");
    } else {
      asm volatile("s_wait_asynccnt 0" ::: "memory");
    }
    __syncthreads();

    // prefetch A a few k-steps ahead (global_prefetch_b8)
    __builtin_prefetch(abase + k0 + 256, 0, 1);

    V16 a;
    a.h[0] = *(const v8h*)(abase + k0 + 8 * g);
    a.h[1] = *(const v8h*)(abase + k0 + 16 + 8 * g);
#pragma unroll
    for (int j = 0; j < 4; ++j) {
      v16h b;  // lane = N column, elements = K (table order)
#pragma unroll
      for (int s = 0; s < 16; ++s) b[s] = sB[cb][8 * g + s + (s & 8)][16 * j + ln];
      acc[j] = wmma_f16(a.v, b, acc[j]);
    }
    __syncthreads();  // all waves done reading before buffer reuse
  }

#pragma unroll
  for (int j = 0; j < 4; ++j) {
    const int col = n0 + 16 * j + ln;
    const float bs = bias[col];
#pragma unroll
    for (int r = 0; r < 8; ++r) {
      const int row = m0 + 16 * wave + r + 8 * g;
      if (row < M) C[(size_t)row * DIM + col] = acc[j][r] + bs;
    }
  }
}

// ---------------------------------------------------------------------------
// RMSNorm (over full 1536) + gain + 3D RoPE; writes f16:
//   Qh[h][tok][d], Vh[h][tok][d], Kt[h][d][tok]  (tok padded to SPAD)
// one block (256 threads) per padded token
// ---------------------------------------------------------------------------
__global__ __launch_bounds__(256) void rms_rope_kernel(
    const float* __restrict__ Qraw, const float* __restrict__ Kraw,
    const float* __restrict__ Vraw, const float* __restrict__ gq,
    const float* __restrict__ gk, const float* __restrict__ freqs,
    _Float16* __restrict__ Qh, _Float16* __restrict__ Kt,
    _Float16* __restrict__ Vh) {
  const int s = blockIdx.x;
  const int t = threadIdx.x;

  if (s >= S_TOT) {  // zero padded tokens (keep NaNs out of WMMA)
    for (int j = t; j < DIM; j += 256) {
      const int n = j >> 7, d = j & 127;
      Qh[((size_t)n * SPAD + s) * HD + d] = (_Float16)0.f;
      Vh[((size_t)n * SPAD + s) * HD + d] = (_Float16)0.f;
      Kt[((size_t)n * HD + d) * SPAD + s] = (_Float16)0.f;
    }
    return;
  }

  __shared__ float red[256];
  const float* qr = Qraw + (size_t)s * DIM;
  const float* kr = Kraw + (size_t)s * DIM;
  float sq = 0.f, sk = 0.f;
  for (int j = t; j < DIM; j += 256) {
    float a = qr[j]; sq += a * a;
    float b = kr[j]; sk += b * b;
  }
  red[t] = sq; __syncthreads();
  for (int off = 128; off > 0; off >>= 1) {
    if (t < off) red[t] += red[t + off];
    __syncthreads();
  }
  const float rq = rsqrtf(red[0] / (float)DIM + 1e-6f);
  __syncthreads();
  red[t] = sk; __syncthreads();
  for (int off = 128; off > 0; off >>= 1) {
    if (t < off) red[t] += red[t + off];
    __syncthreads();
  }
  const float rk = rsqrtf(red[0] / (float)DIM + 1e-6f);

  // token -> (f,h,w) position: F=4, H=20, W=26
  const int fi = s / 520;
  const int rem = s - fi * 520;
  const int hi = rem / 26;
  const int wi = rem - hi * 26;

  // 12 heads x 64 rotary pairs = 768 pairs
  for (int p = t; p < NH * 64; p += 256) {
    const int n = p >> 6, c = p & 63;
    const int idx = (c < 22) ? fi : ((c < 43) ? hi : wi);
    const float ang = freqs[idx * 64 + c];
    float sn, cs;
    __sincosf(ang, &sn, &cs);
    const int base = n * HD + 2 * c;
    const float q0 = qr[base] * rq * gq[base];
    const float q1 = qr[base + 1] * rq * gq[base + 1];
    const float k0 = kr[base] * rk * gk[base];
    const float k1 = kr[base + 1] * rk * gk[base + 1];
    const size_t qo = ((size_t)n * SPAD + s) * HD + 2 * c;
    Qh[qo]     = (_Float16)(q0 * cs - q1 * sn);
    Qh[qo + 1] = (_Float16)(q0 * sn + q1 * cs);
    const size_t ko = ((size_t)n * HD + 2 * c) * SPAD + s;
    Kt[ko]        = (_Float16)(k0 * cs - k1 * sn);
    Kt[ko + SPAD] = (_Float16)(k0 * sn + k1 * cs);
  }

  const float* vr = Vraw + (size_t)s * DIM;
  for (int j = t; j < DIM; j += 256) {
    const int n = j >> 7, d = j & 127;
    Vh[((size_t)n * SPAD + s) * HD + d] = (_Float16)vr[j];
  }
}

// ---------------------------------------------------------------------------
// Flash attention, block-causal (128-token blocks), key padding masked.
// grid = (SPAD/32, NH); block = 64 (2 waves).  Each wave owns 16 q rows.
// Output Oh f16 laid out [token][head*128+d] for the final GEMM.
// ---------------------------------------------------------------------------
__global__ __launch_bounds__(64) void attn_kernel(
    const _Float16* __restrict__ Qh, const _Float16* __restrict__ Kt,
    const _Float16* __restrict__ Vh, _Float16* __restrict__ Oh) {
  __shared__ __align__(32) float    sS[32][128];
  __shared__ __align__(32) float    sO[32][128];
  __shared__ __align__(32) _Float16 sP[32][128];

  const int qblk = blockIdx.x;
  const int h    = blockIdx.y;
  const int tid  = threadIdx.x;
  const int wave = tid >> 5;
  const int lane = tid & 31;
  const int g    = lane >> 4;
  const int ln   = lane & 15;
  const int q0   = qblk * 32;
  const int r0   = wave * 16;

  for (int i = tid; i < 32 * 128; i += 64) (&sO[0][0])[i] = 0.f;
  __syncthreads();

  float m_run = -__builtin_inff();
  float l_run = 0.f;

  const int kbmax = q0 >> 7;  // block-causal at 128 granularity
  const _Float16* Qbase = Qh + ((size_t)h * SPAD + q0) * HD;
  const v8f vzero = {0.f, 0.f, 0.f, 0.f, 0.f, 0.f, 0.f, 0.f};

  for (int kb = 0; kb <= kbmax; ++kb) {
    const int kk_base = kb * 128;

    // ---- S = Q K^T  (16 q rows x 128 keys per wave) ----
    v8f acc[8] = {vzero, vzero, vzero, vzero, vzero, vzero, vzero, vzero};
    for (int d0 = 0; d0 < HD; d0 += 32) {
      V16 a;
      const _Float16* ap = Qbase + (size_t)(r0 + ln) * HD + d0 + 8 * g;
      a.h[0] = *(const v8h*)(ap);
      a.h[1] = *(const v8h*)(ap + 16);
      const _Float16* bp0 = Kt + ((size_t)h * HD + d0) * SPAD + kk_base;
#pragma unroll
      for (int tt = 0; tt < 8; ++tt) {
        v16h b;
#pragma unroll
        for (int s = 0; s < 16; ++s)
          b[s] = bp0[(size_t)(8 * g + s + (s & 8)) * SPAD + 16 * tt + ln];
        acc[tt] = wmma_f16(a.v, b, acc[tt]);
      }
    }
#pragma unroll
    for (int tt = 0; tt < 8; ++tt) {
#pragma unroll
      for (int r = 0; r < 8; ++r) {
        const int row = r0 + r + 8 * g;
        const int col = 16 * tt + ln;
        float v = acc[tt][r] * ATT_SCALE;
        if (kk_base + col >= S_TOT) v = -__builtin_inff();
        sS[row][col] = v;
      }
    }
    __syncthreads();

    // ---- online softmax: thread tid owns q-row tid (tid < 32) ----
    if (tid < 32) {
      float mloc = -__builtin_inff();
      for (int c = 0; c < 128; ++c) mloc = fmaxf(mloc, sS[tid][c]);
      const float mnew = fmaxf(m_run, mloc);
      const float alpha = __expf(m_run - mnew);  // exp(-inf)=0 on first block
      float ssum = 0.f;
      for (int c = 0; c < 128; ++c) {
        const float p = __expf(sS[tid][c] - mnew);
        ssum += p;
        sP[tid][c] = (_Float16)p;
      }
      l_run = l_run * alpha + ssum;
      for (int c = 0; c < 128; ++c) sO[tid][c] *= alpha;
      m_run = mnew;
    }
    __syncthreads();

    // ---- O += P V  (16 q rows x 128 dims per wave) ----
    v8f oacc[8] = {vzero, vzero, vzero, vzero, vzero, vzero, vzero, vzero};
    for (int kk0 = 0; kk0 < 128; kk0 += 32) {
      V16 a;
      const _Float16* ap = &sP[r0 + ln][kk0 + 8 * g];
      a.h[0] = *(const v8h*)(ap);
      a.h[1] = *(const v8h*)(ap + 16);
      const _Float16* bp0 =
          Vh + ((size_t)h * SPAD + kk_base + kk0) * HD;
#pragma unroll
      for (int tt = 0; tt < 8; ++tt) {
        v16h b;
#pragma unroll
        for (int s = 0; s < 16; ++s)
          b[s] = bp0[(size_t)(8 * g + s + (s & 8)) * HD + 16 * tt + ln];
        oacc[tt] = wmma_f16(a.v, b, oacc[tt]);
      }
    }
#pragma unroll
    for (int tt = 0; tt < 8; ++tt) {
#pragma unroll
      for (int r = 0; r < 8; ++r) {
        const int row = r0 + r + 8 * g;
        const int col = 16 * tt + ln;
        sO[row][col] += oacc[tt][r];
      }
    }
    __syncthreads();
  }

  // ---- normalize + store ----
  if (tid < 32) {
    const int token = q0 + tid;
    if (token < S_TOT) {
      const float inv = 1.f / l_run;
      _Float16* op = Oh + (size_t)token * DIM + h * HD;
      for (int c = 0; c < 128; ++c) op[c] = (_Float16)(sO[tid][c] * inv);
    }
  }
}

// ---------------------------------------------------------------------------
// host side
// ---------------------------------------------------------------------------
extern "C" void kernel_launch(void* const* d_in, const int* in_sizes, int n_in,
                              void* d_out, int out_size, void* d_ws,
                              size_t ws_size, hipStream_t stream) {
  (void)in_sizes; (void)n_in; (void)out_size; (void)ws_size;

  const float* x_f   = (const float*)d_in[0];
  const float* freqs = (const float*)d_in[1];
  const float* wq    = (const float*)d_in[2];
  const float* bq    = (const float*)d_in[3];
  const float* wk    = (const float*)d_in[4];
  const float* bk    = (const float*)d_in[5];
  const float* wv    = (const float*)d_in[6];
  const float* bv    = (const float*)d_in[7];
  const float* wo    = (const float*)d_in[8];
  const float* bo    = (const float*)d_in[9];
  const float* gq    = (const float*)d_in[10];
  const float* gk    = (const float*)d_in[11];
  float* out = (float*)d_out;

  char* ws = (char*)d_ws;
  size_t off = 0;
  auto carve = [&](size_t bytes) {
    char* p = ws + off;
    off += (bytes + 255) & ~(size_t)255;
    return p;
  };
  const size_t nXW = (size_t)S_TOT * DIM;      // x / o elements
  const size_t nW  = (size_t)DIM * DIM;        // weight elements
  const size_t nHT = (size_t)NH * SPAD * HD;   // per-head padded tensor

  _Float16* Xh  = (_Float16*)carve(nXW * 2);
  _Float16* Wqh = (_Float16*)carve(nW * 2);
  _Float16* Wkh = (_Float16*)carve(nW * 2);
  _Float16* Wvh = (_Float16*)carve(nW * 2);
  _Float16* Woh = (_Float16*)carve(nW * 2);
  float*    Qr  = (float*)carve(nXW * 4);
  float*    Kr  = (float*)carve(nXW * 4);
  float*    Vr  = (float*)carve(nXW * 4);
  _Float16* Qh  = (_Float16*)carve(nHT * 2);
  _Float16* Kt  = (_Float16*)carve(nHT * 2);
  _Float16* Vh  = (_Float16*)carve(nHT * 2);
  _Float16* Oh  = (_Float16*)carve(nXW * 2);

  // 1) f32 -> f16 conversions
  auto cvt = [&](const float* s, _Float16* d, size_t n) {
    f32_to_f16_kernel<<<(unsigned)((n + 255) / 256), 256, 0, stream>>>(s, d,
                                                                      (int)n);
  };
  cvt(x_f, Xh, nXW);
  cvt(wq, Wqh, nW);
  cvt(wk, Wkh, nW);
  cvt(wv, Wvh, nW);
  cvt(wo, Woh, nW);

  // 2) QKV projections (WMMA GEMM, async-LDS staged B)
  dim3 ggrid((S_TOT + 63) / 64, DIM / 64);
  gemm_f16_kernel<<<ggrid, 128, 0, stream>>>(Xh, Wqh, bq, Qr, S_TOT);
  gemm_f16_kernel<<<ggrid, 128, 0, stream>>>(Xh, Wkh, bk, Kr, S_TOT);
  gemm_f16_kernel<<<ggrid, 128, 0, stream>>>(Xh, Wvh, bv, Vr, S_TOT);

  // 3) RMSNorm + RoPE + layout (and pad-zeroing)
  rms_rope_kernel<<<SPAD, 256, 0, stream>>>(Qr, Kr, Vr, gq, gk, freqs, Qh, Kt,
                                            Vh);

  // 4) flash attention
  dim3 agrid(SPAD / 32, NH);
  attn_kernel<<<agrid, 64, 0, stream>>>(Qh, Kt, Vh, Oh);

  // 5) output projection
  gemm_f16_kernel<<<ggrid, 128, 0, stream>>>(Oh, Woh, bo, out, S_TOT);
}
